// MaxViTEncoderLayer_39917426049201
// MI455X (gfx1250) — compile-verified, matrix operand order checked by reading, not run on hardware
//
#include <hip/hip_runtime.h>
#include <math.h>

#define B_   4
#define L_   6400
#define D_   256
#define H_   8
#define DH_  32
#define C2_  512
#define HW_  80

#define BM 128
#define BN 64
#define BK 32
#define BKP2 40          // BK + 8 bf16 pad: 80B rows -> 16B aligned, bank-spread

typedef __attribute__((ext_vector_type(16))) __bf16 v16bf;
typedef __attribute__((ext_vector_type(8)))  __bf16 v8bf;
typedef __attribute__((ext_vector_type(8)))  float  v8f;

enum { EPI_NONE = 0, EPI_BIAS_RELU = 1, EPI_ELU1 = 2, EPI_ROWSCALE = 3, EPI_BIAS = 4 };

__device__ __forceinline__ __bf16 f2bf(float f) {
  unsigned u = __builtin_bit_cast(unsigned, f);
  u += 0x7FFFu + ((u >> 16) & 1u);                 // RNE
  unsigned short hs = (unsigned short)(u >> 16);
  return __builtin_bit_cast(__bf16, hs);
}

// ---- CDNA5 async global->LDS copy (16B per lane), ASYNCcnt-tracked ----
__device__ __forceinline__ void async_ld_b128(unsigned ldsOff, const void* g) {
  asm volatile("global_load_async_to_lds_b128 %0, %1, off"
               :: "v"(ldsOff), "v"((unsigned long long)(size_t)g)
               : "memory");
}
// SIMM16 immediates: wait for all (0) or allow the 3 just-issued to fly (3).
__device__ __forceinline__ void wait_async_all() {
#if __has_builtin(__builtin_amdgcn_s_wait_asynccnt)
  __builtin_amdgcn_s_wait_asynccnt(0);
#else
  asm volatile("s_wait_asynccnt 0x0" ::: "memory");
#endif
}
__device__ __forceinline__ void wait_async_3() {
#if __has_builtin(__builtin_amdgcn_s_wait_asynccnt)
  __builtin_amdgcn_s_wait_asynccnt(3);
#else
  asm volatile("s_wait_asynccnt 0x3" ::: "memory");
#endif
}

// 16x32 bf16 fragment from LDS tile with K innermost (used for A and for
// pre-transposed B): lane half selects kb=0/8; two contiguous 16B chunks.
__device__ __forceinline__ v16bf load_frag(const __bf16* blk, int ldp, int lane) {
  int m  = lane & 15;
  int kb = (lane & 16) ? 8 : 0;
  const __bf16* p = blk + m * ldp + kb;
  v8bf lo = *(const v8bf*)p;          // K = kb .. kb+7
  v8bf hi = *(const v8bf*)(p + 16);   // K = 16+kb .. 16+kb+7
  return __builtin_shufflevector(lo, hi, 0,1,2,3,4,5,6,7,8,9,10,11,12,13,14,15);
}

__device__ __forceinline__ v8f wmma_bf16(v16bf a, v16bf b, v8f c) {
  return __builtin_amdgcn_wmma_f32_16x16x32_bf16(false, a, false, b, (short)0, c, false, false);
}

// ---------------------------------------------------------------------------
// bf16 WMMA GEMM: C = epi( concatK(A0,A1)[M,K] @ Bt[N,K]^T )
// Double-buffered LDS; tiles staged with async b128 copies (3 per wave per
// tile, uniform so s_wait_asynccnt 3 overlaps copy of tile i+1 with compute
// of tile i). 8 waves, 32x32 WMMA tile each.
// ---------------------------------------------------------------------------
__global__ __launch_bounds__(256)
void gemm_bf16_wmma(const __bf16* __restrict__ A0, const __bf16* __restrict__ A1, int ksplit,
                    const __bf16* __restrict__ Bt, const float* __restrict__ bias,
                    const float* __restrict__ rowScale, void* __restrict__ Cout, int outBf,
                    int M, int N, int K, int lda0, int lda1, int ldbt, int ldc,
                    long aOut, long aIn, long bOut, long bIn, long cOut, long cIn,
                    long sOut, long sIn, int zdiv, int epi) {
  __shared__ __attribute__((aligned(16))) __bf16 As[2][BM][BKP2];
  __shared__ __attribute__((aligned(16))) __bf16 Bs[2][BN][BKP2];

  int z  = blockIdx.z;
  int zo = z / zdiv, zi = z % zdiv;
  A0 += zo * aOut + (long)zi * aIn;
  A1 += zo * aOut + (long)zi * aIn;
  Bt += zo * bOut + (long)zi * bIn;
  long coff = zo * cOut + (long)zi * cIn;
  if (rowScale) rowScale += zo * sOut + (long)zi * sIn;

  int tid  = threadIdx.x;
  int lane = tid & 31;
  int wv   = tid >> 5;
  int wr   = wv & 3;   // wave row (M)
  int wc   = wv >> 2;  // wave col (N)
  int bx   = blockIdx.x, by = blockIdx.y;

  unsigned asBase = (unsigned)(size_t)&As[0][0][0];
  unsigned bsBase = (unsigned)(size_t)&Bs[0][0][0];
  const unsigned asStride = BM * BKP2 * 2;
  const unsigned bsStride = BN * BKP2 * 2;

  int arow = tid >> 2;       // 0..63 (A: 2 passes of 64 rows)
  int ac   = tid & 3;        // 16B chunk in K
  int brow = tid >> 2;       // 0..63 (B: 64 rows, 1 pass)
  int bc   = tid & 3;
  int gcolB = bx * BN + brow; if (gcolB >= N) gcolB = N - 1;  // clamp: uniform count

  // per-thread global source rows (fixed across K loop)
  int growA0 = by * BM + arow;      if (growA0 >= M) growA0 = M - 1;
  int growA1 = by * BM + arow + 64; if (growA1 >= M) growA1 = M - 1;

  // stage one K-tile into LDS buffer `buf` (exactly 3 async insts per wave)
  auto stage = [&](int k0, int buf) {
    int gk = k0 + ac * 8;
    const __bf16* g0 = (gk < ksplit) ? (A0 + (long)growA0 * lda0 + gk)
                                     : (A1 + (long)growA0 * lda1 + (gk - ksplit));
    const __bf16* g1 = (gk < ksplit) ? (A0 + (long)growA1 * lda0 + gk)
                                     : (A1 + (long)growA1 * lda1 + (gk - ksplit));
    unsigned ab = asBase + buf * asStride;
    unsigned bb = bsBase + buf * bsStride;
    async_ld_b128(ab + arow * (BKP2 * 2) + ac * 16, g0);
    async_ld_b128(ab + (arow + 64) * (BKP2 * 2) + ac * 16, g1);
    async_ld_b128(bb + brow * (BKP2 * 2) + bc * 16,
                  Bt + (long)gcolB * ldbt + k0 + bc * 8);
  };

  v8f acc[2][2] = {};

  stage(0, 0);                         // prologue: tile 0 -> buffer 0
  int buf = 0;
  for (int k0 = 0; k0 < K; k0 += BK) {
    bool more = (k0 + BK) < K;
    if (more) stage(k0 + BK, buf ^ 1); // copy tile i+1 under compute of tile i
    if (more) wait_async_3();          // own tile-i copies done; 3 newest in flight
    else      wait_async_all();
    __syncthreads();                   // all waves' tile-i copies visible

    const __bf16* a0 = &As[buf][wr * 32][0];
    const __bf16* b0 = &Bs[buf][wc * 32][0];
    v16bf af0 = load_frag(a0,                 BKP2, lane);
    v16bf af1 = load_frag(a0 + 16 * BKP2,     BKP2, lane);
    v16bf bf0 = load_frag(b0,                 BKP2, lane);
    v16bf bf1 = load_frag(b0 + 16 * BKP2,     BKP2, lane);
    acc[0][0] = wmma_bf16(af0, bf0, acc[0][0]);
    acc[0][1] = wmma_bf16(af0, bf1, acc[0][1]);
    acc[1][0] = wmma_bf16(af1, bf0, acc[1][0]);
    acc[1][1] = wmma_bf16(af1, bf1, acc[1][1]);
    __syncthreads();                   // reads done before this buf is re-staged
    buf ^= 1;
  }

  // ---- epilogue + store (C/D: lanes index N, VGPR r indexes M) ----
  int n_in = lane & 15;
  int mofs = (lane & 16) ? 8 : 0;
  int grb  = by * BM + wr * 32;
  int gcb  = bx * BN + wc * 32;
#pragma unroll
  for (int mi = 0; mi < 2; ++mi) {
#pragma unroll
    for (int ni = 0; ni < 2; ++ni) {
#pragma unroll
      for (int r = 0; r < 8; ++r) {
        int grow = grb + mi * 16 + mofs + r;
        int gcol = gcb + ni * 16 + n_in;
        if (grow < M && gcol < N) {
          float v = acc[mi][ni][r];
          if      (epi == EPI_BIAS_RELU) v = fmaxf(v + bias[gcol], 0.f);
          else if (epi == EPI_ELU1)      v = (v > 0.f) ? v + 1.f : __expf(v);
          else if (epi == EPI_ROWSCALE)  v *= rowScale[grow];
          else if (epi == EPI_BIAS)      v += bias[gcol];
          long idx = coff + (long)grow * ldc + gcol;
          if (outBf) ((__bf16*)Cout)[idx] = f2bf(v);
          else       ((float*)Cout)[idx]  = v;
        }
      }
    }
  }
}

// ---------------------------------------------------------------------------
// Per-(b,h): KV^T (bf16, [dv][dd] for downstream B-operand) and Ksum (f32).
// LDS holds K/V transposed to [dh][s] (K innermost) so A^T/B fragments are
// two contiguous ds_load_b128 per lane. Row pad 8 -> 528B stride, bank-spread.
// ---------------------------------------------------------------------------
__global__ __launch_bounds__(256)
void kv_kernel(const __bf16* __restrict__ Kb, const __bf16* __restrict__ Vb,
               __bf16* __restrict__ KVt, float* __restrict__ Ksum) {
  __shared__ __attribute__((aligned(16))) __bf16 Kst[DH_][264];
  __shared__ __attribute__((aligned(16))) __bf16 Vst[DH_][264];
  __shared__ float redu[8][32][33];
  __shared__ float ksred[8][32];

  int bh = blockIdx.x;
  int b  = bh >> 3, h = bh & 7;
  const __bf16* kb_ = Kb + (long)b * L_ * D_ + h * DH_;
  const __bf16* vb_ = Vb + (long)b * L_ * D_ + h * DH_;

  int tid = threadIdx.x, lane = tid & 31, w = tid >> 5;
  v8f  acc[2][2] = {};
  float kpart = 0.f;

  for (int s0 = 0; s0 < L_; s0 += 256) {
#pragma unroll
    for (int j = 0; j < 4; ++j) {
      int id = tid + 256 * j;          // 1024 chunk tasks: 256 s x 4 d-chunks
      int s  = id & 255;
      int dc = id >> 8;                // 0..3 (8 d's each)
      v8bf kk = *(const v8bf*)(kb_ + (long)(s0 + s) * D_ + dc * 8);
      v8bf vv = *(const v8bf*)(vb_ + (long)(s0 + s) * D_ + dc * 8);
#pragma unroll
      for (int e = 0; e < 8; ++e) {
        Kst[dc * 8 + e][s] = kk[e];
        Vst[dc * 8 + e][s] = vv[e];
      }
    }
    __syncthreads();
#pragma unroll
    for (int j = 0; j < 32; ++j) kpart += (float)Kst[lane][w * 32 + j];

    v16bf af0 = load_frag(&Kst[0][w * 32],  264, lane);   // K^T: dd 0..15
    v16bf af1 = load_frag(&Kst[16][w * 32], 264, lane);   // K^T: dd 16..31
    v16bf bf0 = load_frag(&Vst[0][w * 32],  264, lane);
    v16bf bf1 = load_frag(&Vst[16][w * 32], 264, lane);
    acc[0][0] = wmma_bf16(af0, bf0, acc[0][0]);
    acc[0][1] = wmma_bf16(af0, bf1, acc[0][1]);
    acc[1][0] = wmma_bf16(af1, bf0, acc[1][0]);
    acc[1][1] = wmma_bf16(af1, bf1, acc[1][1]);
    __syncthreads();
  }

  int n_in = lane & 15, mofs = (lane & 16) ? 8 : 0;
#pragma unroll
  for (int mi = 0; mi < 2; ++mi)
#pragma unroll
    for (int ni = 0; ni < 2; ++ni)
#pragma unroll
      for (int r = 0; r < 8; ++r)
        redu[w][mi * 16 + mofs + r][ni * 16 + n_in] = acc[mi][ni][r];
  ksred[w][lane] = kpart;
  __syncthreads();

  for (int i = tid; i < DH_ * DH_; i += 256) {
    int dd = i >> 5, dv = i & 31;
    float s = 0.f;
#pragma unroll
    for (int j = 0; j < 8; ++j) s += redu[j][dd][dv];
    KVt[(long)bh * DH_ * DH_ + (long)dv * DH_ + dd] = f2bf(s);   // transposed
  }
  if (tid < DH_) {
    float s = 0.f;
#pragma unroll
    for (int j = 0; j < 8; ++j) s += ksred[j][tid];
    Ksum[bh * DH_ + tid] = s;
  }
}

// Z[b,h,l] = 1 / (dot(Q[b,l,h,:], Ksum[b,h,:]) + eps)
__global__ __launch_bounds__(256)
void z_kernel(const __bf16* __restrict__ Qb, const float* __restrict__ Ksum,
              float* __restrict__ Z) {
  long idx = (long)blockIdx.x * 256 + threadIdx.x;
  if (idx >= (long)B_ * H_ * L_) return;
  int  l  = (int)(idx % L_);
  int  bh = (int)(idx / L_);
  int  h  = bh & 7, b = bh >> 3;
  const v8bf* q8 = (const v8bf*)(Qb + ((long)b * L_ + l) * D_ + h * DH_);
  const float* ks = Ksum + bh * DH_;
  float s = 0.f;
#pragma unroll
  for (int c = 0; c < 4; ++c) {
    v8bf q = q8[c];
#pragma unroll
    for (int e = 0; e < 8; ++e) s += (float)q[e] * ks[c * 8 + e];
  }
  Z[idx] = 1.0f / (s + 1e-6f);
}

// Depthwise 3x3 (pad 1) + bias + exact GELU; bf16 in/out, f32 accumulate.
__global__ __launch_bounds__(256)
void dwconv_gelu(const __bf16* __restrict__ y, const float* __restrict__ wgt,
                 const float* __restrict__ bias, __bf16* __restrict__ out) {
  long idx = (long)blockIdx.x * 256 + threadIdx.x;
  if (idx >= (long)B_ * L_ * C2_) return;
  int  c  = (int)(idx % C2_);
  long bl = idx / C2_;
  int  l  = (int)(bl % L_);
  int  b  = (int)(bl / L_);
  int  hh = l / HW_, ww = l % HW_;
  float acc = bias[c];
  const float* w9 = wgt + c * 9;
#pragma unroll
  for (int dy = -1; dy <= 1; ++dy) {
    int yy = hh + dy;
    if (yy < 0 || yy >= HW_) continue;
#pragma unroll
    for (int dx = -1; dx <= 1; ++dx) {
      int xx = ww + dx;
      if (xx < 0 || xx >= HW_) continue;
      acc += (float)y[((long)b * L_ + yy * HW_ + xx) * C2_ + c] * w9[(dy + 1) * 3 + (dx + 1)];
    }
  }
  out[idx] = f2bf(0.5f * acc * (1.0f + erff(acc * 0.70710678118654752f)));
}

// Wave-per-token LayerNorm over D=256; optional residual; f32 or bf16 out.
__global__ __launch_bounds__(256)
void ln_res(const float* __restrict__ in, const float* __restrict__ g,
            const float* __restrict__ b, const float* __restrict__ res,
            void* __restrict__ out, int outBf) {
  int token = blockIdx.x * 8 + (threadIdx.x >> 5);
  int lane  = threadIdx.x & 31;
  if (token >= B_ * L_) return;
  const float* p = in + (long)token * D_;
  float v[8];
  float s = 0.f;
#pragma unroll
  for (int i = 0; i < 8; ++i) { v[i] = p[lane + 32 * i]; s += v[i]; }
#pragma unroll
  for (int o = 16; o; o >>= 1) s += __shfl_xor(s, o, 32);
  float mu = s * (1.0f / D_);
  float q = 0.f;
#pragma unroll
  for (int i = 0; i < 8; ++i) { float d = v[i] - mu; q += d * d; }
#pragma unroll
  for (int o = 16; o; o >>= 1) q += __shfl_xor(q, o, 32);
  float inv = rsqrtf(q * (1.0f / D_) + 1e-5f);
#pragma unroll
  for (int i = 0; i < 8; ++i) {
    int col = lane + 32 * i;
    float r = (v[i] - mu) * inv * g[col] + b[col];
    if (res) r += res[(long)token * D_ + col];
    long off = (long)token * D_ + col;
    if (outBf) ((__bf16*)out)[off] = f2bf(r);
    else       ((float*)out)[off]  = r;
  }
}

// fp32 -> bf16 (flat) and fp32 -> bf16 transposed (for B operands).
__global__ __launch_bounds__(256)
void cvt_bf16(const float* __restrict__ in, __bf16* __restrict__ out, long n) {
  long i = ((long)blockIdx.x * 256 + threadIdx.x) * 4;
  if (i >= n) return;
  float4 v = *(const float4*)&in[i];
  out[i + 0] = f2bf(v.x); out[i + 1] = f2bf(v.y);
  out[i + 2] = f2bf(v.z); out[i + 3] = f2bf(v.w);
}
__global__ __launch_bounds__(256)
void cvt_tr_bf16(const float* __restrict__ in, __bf16* __restrict__ out, int K, int N) {
  long i = (long)blockIdx.x * 256 + threadIdx.x;
  if (i >= (long)K * N) return;
  int n = (int)(i % N), k = (int)(i / N);
  out[(long)n * K + k] = f2bf(in[i]);
}

extern "C" void kernel_launch(void* const* d_in, const int* in_sizes, int n_in,
                              void* d_out, int out_size, void* d_ws, size_t ws_size,
                              hipStream_t stream) {
  const float* x    = (const float*)d_in[0];
  const float* src  = (const float*)d_in[1];
  const float* Wq   = (const float*)d_in[2];
  const float* Wk   = (const float*)d_in[3];
  const float* Wv   = (const float*)d_in[4];
  const float* Wm   = (const float*)d_in[5];
  const float* ln1g = (const float*)d_in[6];
  const float* ln1b = (const float*)d_in[7];
  const float* fc1w = (const float*)d_in[8];
  const float* fc1b = (const float*)d_in[9];
  const float* dww  = (const float*)d_in[10];
  const float* dwb  = (const float*)d_in[11];
  const float* fc2w = (const float*)d_in[12];
  const float* fc2b = (const float*)d_in[13];
  const float* ln2g = (const float*)d_in[14];
  const float* ln2b = (const float*)d_in[15];
  float* out = (float*)d_out;
  (void)in_sizes; (void)n_in; (void)out_size; (void)ws_size;

  const long M = (long)B_ * L_;  // 25600 tokens
  char* wp = (char*)d_ws;
  auto carve = [&](size_t bytes) { char* p = wp; wp += (bytes + 255) & ~(size_t)255; return p; };
  __bf16* xb    = (__bf16*)carve(M * D_ * 2);
  __bf16* srcb  = (__bf16*)carve(M * D_ * 2);
  __bf16* Wqt   = (__bf16*)carve((size_t)D_ * D_ * 2);
  __bf16* Wkt   = (__bf16*)carve((size_t)D_ * D_ * 2);
  __bf16* Wvt   = (__bf16*)carve((size_t)D_ * D_ * 2);
  __bf16* Wmt   = (__bf16*)carve((size_t)D_ * D_ * 2);
  __bf16* fc1wt = (__bf16*)carve((size_t)C2_ * C2_ * 2);
  __bf16* fc2wt = (__bf16*)carve((size_t)C2_ * D_ * 2);
  __bf16* Qb    = (__bf16*)carve(M * D_ * 2);
  __bf16* Kb    = (__bf16*)carve(M * D_ * 2);   // later reused as msg (bf16)
  __bf16* Vb    = (__bf16*)carve(M * D_ * 2);   // later reused as msg0 (bf16)
  float*  tmpF  = (float*)carve(M * D_ * 4);    // Wm out, then fc2 out
  __bf16* y1b   = (__bf16*)carve(M * C2_ * 2);
  __bf16* y2b   = (__bf16*)carve(M * C2_ * 2);
  __bf16* KVt   = (__bf16*)carve((size_t)B_ * H_ * DH_ * DH_ * 2);
  float*  Ksum  = (float*)carve((size_t)B_ * H_ * DH_ * 4);
  float*  Zb    = (float*)carve((size_t)B_ * H_ * L_ * 4);

  dim3 blk(256);
  const int BIG = 1 << 30;

  // 0) precision/layout pre-passes: activations flat bf16, weights bf16 [N][K]
  cvt_bf16<<<dim3((int)(M * D_ / 1024)), blk, 0, stream>>>(x, xb, M * D_);
  cvt_bf16<<<dim3((int)(M * D_ / 1024)), blk, 0, stream>>>(src, srcb, M * D_);
  cvt_tr_bf16<<<dim3(D_ * D_ / 256), blk, 0, stream>>>(Wq, Wqt, D_, D_);
  cvt_tr_bf16<<<dim3(D_ * D_ / 256), blk, 0, stream>>>(Wk, Wkt, D_, D_);
  cvt_tr_bf16<<<dim3(D_ * D_ / 256), blk, 0, stream>>>(Wv, Wvt, D_, D_);
  cvt_tr_bf16<<<dim3(D_ * D_ / 256), blk, 0, stream>>>(Wm, Wmt, D_, D_);
  cvt_tr_bf16<<<dim3(C2_ * C2_ / 256), blk, 0, stream>>>(fc1w, fc1wt, C2_, C2_);
  cvt_tr_bf16<<<dim3(C2_ * D_ / 256), blk, 0, stream>>>(fc2w, fc2wt, C2_, D_);

  // 1-3) q/k/v projections (q,k fused elu+1)
  gemm_bf16_wmma<<<dim3(D_ / BN, (int)(M / BM), 1), blk, 0, stream>>>(
      xb, xb, BIG, Wqt, nullptr, nullptr, Qb, 1, (int)M, D_, D_, D_, D_, D_, D_,
      0, 0, 0, 0, 0, 0, 0, 0, 1, EPI_ELU1);
  gemm_bf16_wmma<<<dim3(D_ / BN, (int)(M / BM), 1), blk, 0, stream>>>(
      srcb, srcb, BIG, Wkt, nullptr, nullptr, Kb, 1, (int)M, D_, D_, D_, D_, D_, D_,
      0, 0, 0, 0, 0, 0, 0, 0, 1, EPI_ELU1);
  gemm_bf16_wmma<<<dim3(D_ / BN, (int)(M / BM), 1), blk, 0, stream>>>(
      srcb, srcb, BIG, Wvt, nullptr, nullptr, Vb, 1, (int)M, D_, D_, D_, D_, D_, D_,
      0, 0, 0, 0, 0, 0, 0, 0, 1, EPI_NONE);

  // 4-5) per-head KV^T + Ksum, then Z normalizers
  kv_kernel<<<dim3(B_ * H_), blk, 0, stream>>>(Kb, Vb, KVt, Ksum);
  z_kernel<<<dim3((int)(((long)B_ * H_ * L_ + 255) / 256)), blk, 0, stream>>>(Qb, Ksum, Zb);

  // 6) msg0 = (Q @ KV) * Z, batched over z = b*H+h (overwrites Vb)
  gemm_bf16_wmma<<<dim3(1, L_ / BM, B_ * H_), blk, 0, stream>>>(
      Qb, Qb, BIG, KVt, nullptr, Zb, Vb, 1, L_, DH_, DH_, D_, D_, DH_, D_,
      (long)L_ * D_, DH_, (long)H_ * DH_ * DH_, (long)DH_ * DH_,
      (long)L_ * D_, DH_, (long)H_ * L_, L_, H_, EPI_ROWSCALE);

  // 7-8) msg0 @ Wm -> f32, LN1 -> msg (bf16, into Kb)
  gemm_bf16_wmma<<<dim3(D_ / BN, (int)(M / BM), 1), blk, 0, stream>>>(
      Vb, Vb, BIG, Wmt, nullptr, nullptr, tmpF, 0, (int)M, D_, D_, D_, D_, D_, D_,
      0, 0, 0, 0, 0, 0, 0, 0, 1, EPI_NONE);
  ln_res<<<dim3((int)(M / 8)), blk, 0, stream>>>(tmpF, ln1g, ln1b, nullptr, Kb, 1);

  // 9) fc1 on concat([x, msg]) + bias + ReLU -> y1 (bf16)
  gemm_bf16_wmma<<<dim3(C2_ / BN, (int)(M / BM), 1), blk, 0, stream>>>(
      xb, Kb, D_, fc1wt, fc1b, nullptr, y1b, 1, (int)M, C2_, C2_, D_, D_, C2_, C2_,
      0, 0, 0, 0, 0, 0, 0, 0, 1, EPI_BIAS_RELU);

  // 10) depthwise 3x3 + bias + GELU -> y2 (bf16)
  dwconv_gelu<<<dim3((int)((M * C2_ + 255) / 256)), blk, 0, stream>>>(y1b, dww, dwb, y2b);

  // 11-12) fc2 + bias -> f32, LN2 + residual -> out
  gemm_bf16_wmma<<<dim3(D_ / BN, (int)(M / BM), 1), blk, 0, stream>>>(
      y2b, y2b, BIG, fc2wt, fc2b, nullptr, tmpF, 0, (int)M, D_, C2_, C2_, C2_, C2_, D_,
      0, 0, 0, 0, 0, 0, 0, 0, 1, EPI_BIAS);
  ln_res<<<dim3((int)(M / 8)), blk, 0, stream>>>(tmpF, ln2g, ln2b, x, out, 0);
}
